// TransformerEncoderLayer_74491912781841
// MI455X (gfx1250) — matrix operand order repaired and checked by reading, not compiled, not run on hardware
//
#include <hip/hip_runtime.h>

// ---------------------------------------------------------------------------
// Types
// ---------------------------------------------------------------------------
typedef __attribute__((ext_vector_type(16))) __bf16        v16bf;
typedef __attribute__((ext_vector_type(8)))  float         v8f;
typedef __attribute__((ext_vector_type(4)))  unsigned int  u32x4;

union Frag16 { v16bf v; u32x4 u[2]; };

// ---------------------------------------------------------------------------
// Problem constants (match reference)
// ---------------------------------------------------------------------------
#define HH   4
#define DD   256
#define KD   1024
#define FFD  1024
#define LL   2048
#define BB   4
#define LB   (LL * BB)      // 8192 rows of (l,b)

// GEMM tiling
#define BM 128
#define BN 128
#define BK 32
#define PITCH 40            // bf16 elems per LDS row (80B, multiple of 16B)
#define ABUF  (BM * PITCH)  // elems per A LDS buffer
#define BBUF  (BN * PITCH)

#define FLAG_RELU 1

// ---------------------------------------------------------------------------
// Generic bf16 WMMA GEMM:  C[M,N] = A[M,Kd] * B[N,Kd]^T (+bias[N]) (+resid) (relu)
// A, B row-major with arbitrary row strides (elements). Optional fp32 and bf16
// outputs (either may be null). M % 128 == 0, N % 128 == 0, Kd % 32 == 0.
//
// Data movement: CDNA5 async DMA global->LDS (ASYNCcnt) with double-buffered
// LDS ping-pong so the transfer of tile k+1 overlaps the WMMAs on tile k.
// ---------------------------------------------------------------------------
__launch_bounds__(256)
__global__ void gemm_bf16(const __bf16* __restrict__ A, long long lda,
                          const __bf16* __restrict__ B, long long ldb,
                          float* __restrict__ C, long long ldc,
                          __bf16* __restrict__ Cb, long long ldcb,
                          const float* __restrict__ bias,
                          const float* __restrict__ resid, long long ldr,
                          int Kd, int flags)
{
    __shared__ __attribute__((aligned(16))) __bf16 As[2 * ABUF];
    __shared__ __attribute__((aligned(16))) __bf16 Bs[2 * BBUF];

    const int tid  = threadIdx.x;
    const int lane = tid & 31;
    const int wave = tid >> 5;        // 0..7
    const int wm   = wave >> 1;       // 0..3 : 32-row slab
    const int wn   = wave & 1;        // 0..1 : 64-col slab
    const int l15  = lane & 15;
    const int sel  = lane >> 4;       // half-wave select

    const long long rowM0 = (long long)blockIdx.y * BM;
    const long long colN0 = (long long)blockIdx.x * BN;

    // global->LDS staging: each thread moves 32B of A and 32B of B per tile
    const int ldRow  = tid >> 1;          // 0..127
    const int ldHalf = (tid & 1) << 4;    // 0 or 16 elems

    const __bf16* Aptr = A + (rowM0 + ldRow) * lda + ldHalf;
    const __bf16* Bptr = B + (colN0 + ldRow) * ldb + ldHalf;

    // Async DMA of one 32x(BM+BN) bf16 K-tile into LDS buffer `buf`.
    // 4 x GLOBAL_LOAD_ASYNC_TO_LDS_B128 per thread; LDS and global deltas for
    // the second 16B chunk are both +16B, expressed via the inst offset.
    auto issue_tile = [&](int k0, int buf) {
        const unsigned long long gA =
            (unsigned long long)(uintptr_t)(Aptr + k0);
        const unsigned long long gB =
            (unsigned long long)(uintptr_t)(Bptr + k0);
        const unsigned la =
            (unsigned)(uintptr_t)&As[buf * ABUF + ldRow * PITCH + ldHalf];
        const unsigned lb =
            (unsigned)(uintptr_t)&Bs[buf * BBUF + ldRow * PITCH + ldHalf];
        asm volatile("global_load_async_to_lds_b128 %0, %1, off"
                     :: "v"(la), "v"(gA) : "memory");
        asm volatile("global_load_async_to_lds_b128 %0, %1, off offset:16"
                     :: "v"(la), "v"(gA) : "memory");
        asm volatile("global_load_async_to_lds_b128 %0, %1, off"
                     :: "v"(lb), "v"(gB) : "memory");
        asm volatile("global_load_async_to_lds_b128 %0, %1, off offset:16"
                     :: "v"(lb), "v"(gB) : "memory");
    };

    v8f acc[2][4] = {};

    const int nk = Kd / BK;
    issue_tile(0, 0);
    if (nk > 1) issue_tile(BK, 1);

    for (int ks = 0; ks < nk; ++ks) {
        const int buf = ks & 1;

        // Tile ks complete when at most the (optional) tile ks+1 group (4 ops)
        // remains outstanding; async loads retire in order per wave.
        if (ks + 1 < nk) asm volatile("s_wait_asynccnt 0x4" ::: "memory");
        else             asm volatile("s_wait_asynccnt 0x0" ::: "memory");
        __syncthreads();   // all waves' DMA done -> tile visible in LDS

        Frag16 a[2], b[4];
#pragma unroll
        for (int mt = 0; mt < 2; ++mt) {
            // 16-bit A 16x32 layout: lanes<16 -> K 0..7/16..23, lanes>=16 -> +8
            const __bf16* ar = &As[buf * ABUF + (wm * 32 + mt * 16 + l15) * PITCH];
            a[mt].u[0] = *(const u32x4*)(ar + sel * 8);
            a[mt].u[1] = *(const u32x4*)(ar + 16 + sel * 8);
        }
#pragma unroll
        for (int nt = 0; nt < 4; ++nt) {
            // 16-bit B 32x16 layout: lane = N, lanes<16 -> K 0..7/16..23, lanes>=16 -> +8
            const __bf16* br =
                &Bs[buf * BBUF + (wn * 64 + nt * 16 + l15) * PITCH + sel * 8];
            b[nt].u[0] = *(const u32x4*)(br);
            b[nt].u[1] = *(const u32x4*)(br + 16);
        }
#pragma unroll
        for (int mt = 0; mt < 2; ++mt)
#pragma unroll
            for (int nt = 0; nt < 4; ++nt)
                acc[mt][nt] = __builtin_amdgcn_wmma_f32_16x16x32_bf16(
                    false, a[mt].v, false, b[nt].v, (short)0, acc[mt][nt],
                    false, false);

        if (ks + 2 < nk) {
            __syncthreads();              // everyone done reading buffer `buf`
            issue_tile((ks + 2) * BK, buf);  // refill it for tile ks+2
        }
    }

    // Epilogue. 32-bit C/D layout: lanes<16 -> M = vgpr, lanes>=16 -> M = 8+vgpr
#pragma unroll
    for (int mt = 0; mt < 2; ++mt) {
#pragma unroll
        for (int nt = 0; nt < 4; ++nt) {
            const long long row0 = rowM0 + wm * 32 + mt * 16 + sel * 8;
            const long long col  = colN0 + wn * 64 + nt * 16 + l15;
            const float bv = bias ? bias[col] : 0.0f;
            v8f c = acc[mt][nt];
#pragma unroll
            for (int r = 0; r < 8; ++r) {
                const long long row = row0 + r;
                float v = c[r] + bv;
                if (resid) v += resid[row * ldr + col];
                if (flags & FLAG_RELU) v = fmaxf(v, 0.0f);
                if (C)  C[row * ldc + col]   = v;
                if (Cb) Cb[row * ldcb + col] = (__bf16)v;
            }
        }
    }
}

// ---------------------------------------------------------------------------
// Elementwise / rowwise helper kernels (fp32 math)
// ---------------------------------------------------------------------------
__global__ void f32_to_bf16(const float* __restrict__ in,
                            __bf16* __restrict__ out, int n)
{
    int i = blockIdx.x * blockDim.x + threadIdx.x;
    if (i < n) out[i] = (__bf16)in[i];
}

// out_bf16 = in / max(||row||_2, 1e-12); one block per row
__global__ void l2norm_rows(const float* __restrict__ in, long long ldin,
                            __bf16* __restrict__ out, long long ldout, int ncols)
{
    __shared__ float red[256];
    const long long row = blockIdx.x;
    const float* r = in + row * ldin;
    float s = 0.0f;
    for (int c = threadIdx.x; c < ncols; c += 256) { float v = r[c]; s += v * v; }
    red[threadIdx.x] = s; __syncthreads();
    for (int st = 128; st > 0; st >>= 1) {
        if (threadIdx.x < st) red[threadIdx.x] += red[threadIdx.x + st];
        __syncthreads();
    }
    const float inv = 1.0f / fmaxf(sqrtf(red[0]), 1e-12f);
    for (int c = threadIdx.x; c < ncols; c += 256)
        out[row * ldout + c] = (__bf16)(r[c] * inv);
}

// in-place row softmax; one block per row
__global__ void softmax_rows(float* __restrict__ S, int ncols)
{
    __shared__ float red[256];
    float* r = S + (long long)blockIdx.x * ncols;
    float m = -3.4e38f;
    for (int c = threadIdx.x; c < ncols; c += 256) m = fmaxf(m, r[c]);
    red[threadIdx.x] = m; __syncthreads();
    for (int st = 128; st > 0; st >>= 1) {
        if (threadIdx.x < st) red[threadIdx.x] = fmaxf(red[threadIdx.x], red[threadIdx.x + st]);
        __syncthreads();
    }
    m = red[0]; __syncthreads();
    float s = 0.0f;
    for (int c = threadIdx.x; c < ncols; c += 256) {
        float e = __expf(r[c] - m);
        r[c] = e; s += e;
    }
    red[threadIdx.x] = s; __syncthreads();
    for (int st = 128; st > 0; st >>= 1) {
        if (threadIdx.x < st) red[threadIdx.x] += red[threadIdx.x + st];
        __syncthreads();
    }
    const float inv = 1.0f / red[0];
    for (int c = threadIdx.x; c < ncols; c += 256) r[c] *= inv;
}

__global__ void colsum(const float* __restrict__ S, float* __restrict__ cs,
                       int rows, int ncols)
{
    int col = blockIdx.x * blockDim.x + threadIdx.x;
    float s = 0.0f;
    for (int l = 0; l < rows; ++l) s += S[(long long)l * ncols + col];
    cs[col] = s;
}

// P_bf16 = S / (1e-9 + colsum[col])
__global__ void renorm_bf16(const float* __restrict__ S,
                            const float* __restrict__ cs,
                            __bf16* __restrict__ P, int ncols)
{
    long long i = (long long)blockIdx.x * blockDim.x + threadIdx.x;
    int col = (int)(i % ncols);
    P[i] = (__bf16)(S[i] / (1e-9f + cs[col]));
}

// wv (LB x D, row = l*BB+b) -> wvT[((b*D+e)*LL)+l] bf16 (one head)
__global__ void wv_transpose(const float* __restrict__ wv, __bf16* __restrict__ wvT)
{
    int i = blockIdx.x * blockDim.x + threadIdx.x;   // over LB*DD
    int e  = i % DD;
    int rb = i / DD;
    int b  = rb % BB;
    int l  = rb / BB;
    wvT[((long long)(b * DD + e)) * LL + l] = (__bf16)wv[i];
}

__global__ void diff_bf16(const float* __restrict__ a, const float* __restrict__ b,
                          __bf16* __restrict__ out, int n)
{
    int i = blockIdx.x * blockDim.x + threadIdx.x;
    if (i < n) out[i] = (__bf16)(a[i] - b[i]);
}

// LayerNorm over 256 cols; one block (256 thr) per row; optional fp32/bf16 out
__global__ void layernorm256(const float* __restrict__ in, long long ldin,
                             const float* __restrict__ g, const float* __restrict__ bta,
                             float* __restrict__ o32, long long ldo,
                             __bf16* __restrict__ ob, long long ldob)
{
    __shared__ float red[256];
    const int c = threadIdx.x;
    const long long row = blockIdx.x;
    const float v = in[row * ldin + c];
    red[c] = v; __syncthreads();
    for (int st = 128; st > 0; st >>= 1) {
        if (c < st) red[c] += red[c + st];
        __syncthreads();
    }
    const float mean = red[0] * (1.0f / 256.0f);
    __syncthreads();
    const float d = v - mean;
    red[c] = d * d; __syncthreads();
    for (int st = 128; st > 0; st >>= 1) {
        if (c < st) red[c] += red[c + st];
        __syncthreads();
    }
    const float var = red[0] * (1.0f / 256.0f);
    const float y = d * rsqrtf(var + 1e-5f) * g[c] + bta[c];
    if (o32) o32[row * ldo + c] = y;
    if (ob)  ob[row * ldob + c] = (__bf16)y;
}

// ---------------------------------------------------------------------------
// Host orchestration
// ---------------------------------------------------------------------------
static inline void launch_gemm(hipStream_t s,
                               const __bf16* A, long long lda,
                               const __bf16* B, long long ldb,
                               float* C, long long ldc,
                               __bf16* Cb, long long ldcb,
                               const float* bias,
                               const float* resid, long long ldr,
                               int M, int N, int Kd, int flags)
{
    dim3 grid(N / BN, M / BM), blk(256);
    gemm_bf16<<<grid, blk, 0, s>>>(A, lda, B, ldb, C, ldc, Cb, ldcb,
                                   bias, resid, ldr, Kd, flags);
}

extern "C" void kernel_launch(void* const* d_in, const int* in_sizes, int n_in,
                              void* d_out, int out_size, void* d_ws, size_t ws_size,
                              hipStream_t stream)
{
    (void)in_sizes; (void)n_in; (void)out_size; (void)ws_size;

    const float* src    = (const float*)d_in[0];   // (L,B,D)
    const float* WK     = (const float*)d_in[1];   // (H,K,D)
    const float* WQ     = (const float*)d_in[2];   // (H,K,D)
    const float* WV     = (const float*)d_in[3];   // (H,D,D)
    const float* TC     = (const float*)d_in[4];   // (H,D,D)
    const float* ENw1   = (const float*)d_in[5];   // (H,FF,D)
    const float* ENb1   = (const float*)d_in[6];   // (H,FF)
    const float* ENw2   = (const float*)d_in[7];   // (H,D,FF)
    const float* ENb2   = (const float*)d_in[8];   // (H,D)
    const float* ENg    = (const float*)d_in[9];   // (H,D)
    const float* ENbeta = (const float*)d_in[10];  // (H,D)
    const float* Wout   = (const float*)d_in[11];  // (D,H*D)
    const float* L1w    = (const float*)d_in[12];  // (FF,D)
    const float* L1b    = (const float*)d_in[13];
    const float* L2w    = (const float*)d_in[14];  // (D,FF)
    const float* L2b    = (const float*)d_in[15];
    const float* N1g    = (const float*)d_in[16];
    const float* N1b    = (const float*)d_in[17];
    const float* N2g    = (const float*)d_in[18];
    const float* N2b    = (const float*)d_in[19];
    float* outp = (float*)d_out;

    // ---- carve workspace -------------------------------------------------
    size_t off = 0;
    char* base = (char*)d_ws;
    auto carve = [&](size_t bytes) -> void* {
        off = (off + 255) & ~(size_t)255;
        void* p = base + off;
        off += bytes;
        return p;
    };
    const size_t B2 = sizeof(__bf16);

    __bf16* srcb   = (__bf16*)carve((size_t)LB * DD * B2);
    __bf16* WQb    = (__bf16*)carve((size_t)HH * KD * DD * B2);
    __bf16* WKb    = (__bf16*)carve((size_t)HH * KD * DD * B2);
    __bf16* WVb    = (__bf16*)carve((size_t)HH * DD * DD * B2);
    __bf16* TCb    = (__bf16*)carve((size_t)HH * DD * DD * B2);
    __bf16* ENw1b  = (__bf16*)carve((size_t)HH * FFD * DD * B2);
    __bf16* ENw2b  = (__bf16*)carve((size_t)HH * DD * FFD * B2);
    __bf16* Woutb  = (__bf16*)carve((size_t)DD * HH * DD * B2);
    __bf16* L1wb   = (__bf16*)carve((size_t)FFD * DD * B2);
    __bf16* L2wb   = (__bf16*)carve((size_t)DD * FFD * B2);

    float*  scr32  = (float*)carve((size_t)LB * KD * 4);        // gemm fp32 scratch
    __bf16* wqb    = (__bf16*)carve((size_t)HH * LB * KD * B2); // (h, l*B+b, k)
    __bf16* wkb    = (__bf16*)carve((size_t)HH * LB * KD * B2);
    __bf16* wvTb   = (__bf16*)carve((size_t)HH * BB * DD * LL * B2); // (h,b,e,l)
    float*  S      = (float*)carve((size_t)LL * LL * 4);        // per (h,b) scores
    float*  csum   = (float*)carve((size_t)LL * 4);
    __bf16* Pb     = (__bf16*)carve((size_t)LL * LL * B2);
    float*  out32  = (float*)carve((size_t)HH * LB * DD * 4);   // (h, l*B+b, e)
    __bf16* diffb  = (__bf16*)carve((size_t)HH * LB * DD * B2);
    float*  ru32   = (float*)carve((size_t)HH * LB * DD * 4);
    __bf16* rub    = (__bf16*)carve((size_t)HH * LB * DD * B2);
    __bf16* hidb   = (__bf16*)carve((size_t)LB * FFD * B2);     // reused
    float*  sum32  = (float*)carve((size_t)LB * DD * 4);        // reused
    __bf16* concb  = (__bf16*)carve((size_t)LB * HH * DD * B2);
    float*  x32    = (float*)carve((size_t)LB * DD * 4);
    __bf16* xb     = (__bf16*)carve((size_t)LB * DD * B2);

    // ---- weight / activation conversion ---------------------------------
    auto cvt = [&](const float* in, __bf16* ot, int n) {
        f32_to_bf16<<<n / 256, 256, 0, stream>>>(in, ot, n);
    };
    cvt(src,  srcb,  LB * DD);
    cvt(WQ,   WQb,   HH * KD * DD);
    cvt(WK,   WKb,   HH * KD * DD);
    cvt(WV,   WVb,   HH * DD * DD);
    cvt(TC,   TCb,   HH * DD * DD);
    cvt(ENw1, ENw1b, HH * FFD * DD);
    cvt(ENw2, ENw2b, HH * DD * FFD);
    cvt(Wout, Woutb, DD * HH * DD);
    cvt(L1w,  L1wb,  FFD * DD);
    cvt(L2w,  L2wb,  DD * FFD);

    // ---- per-head projections + attention -------------------------------
    for (int h = 0; h < HH; ++h) {
        // wq = l2norm(src @ WQ[h]^T)
        launch_gemm(stream, srcb, DD, WQb + (size_t)h * KD * DD, DD,
                    scr32, KD, nullptr, 0, nullptr, nullptr, 0, LB, KD, DD, 0);
        l2norm_rows<<<LB, 256, 0, stream>>>(scr32, KD,
                                            wqb + (size_t)h * LB * KD, KD, KD);
        // wk = l2norm(src @ WK[h]^T)
        launch_gemm(stream, srcb, DD, WKb + (size_t)h * KD * DD, DD,
                    scr32, KD, nullptr, 0, nullptr, nullptr, 0, LB, KD, DD, 0);
        l2norm_rows<<<LB, 256, 0, stream>>>(scr32, KD,
                                            wkb + (size_t)h * LB * KD, KD, KD);
        // wv = src @ WV[h]^T, then transpose to (b,e,l) bf16
        launch_gemm(stream, srcb, DD, WVb + (size_t)h * DD * DD, DD,
                    scr32, DD, nullptr, 0, nullptr, nullptr, 0, LB, DD, DD, 0);
        wv_transpose<<<(LB * DD) / 256, 256, 0, stream>>>(
            scr32, wvTb + (size_t)h * BB * DD * LL);

        for (int b = 0; b < BB; ++b) {
            const __bf16* Qhb = wqb + (size_t)h * LB * KD + (size_t)b * KD;
            const __bf16* Khb = wkb + (size_t)h * LB * KD + (size_t)b * KD;
            // S = Q K^T  (2048 x 2048 x 1024)
            launch_gemm(stream, Qhb, (long long)BB * KD, Khb, (long long)BB * KD,
                        S, LL, nullptr, 0, nullptr, nullptr, 0, LL, LL, KD, 0);
            softmax_rows<<<LL, 256, 0, stream>>>(S, LL);
            colsum<<<LL / 256, 256, 0, stream>>>(S, csum, LL, LL);
            renorm_bf16<<<(LL * LL) / 256, 256, 0, stream>>>(S, csum, Pb, LL);
            // out[l,e] = P @ wvT^T  -> written to out32 (h, l*B+b, e)
            launch_gemm(stream, Pb, LL,
                        wvTb + ((size_t)(h * BB + b)) * DD * LL, LL,
                        out32 + (size_t)h * LB * DD + (size_t)b * DD,
                        (long long)BB * DD,
                        nullptr, 0, nullptr, nullptr, 0, LL, DD, LL, 0);
        }
    }

    // ---- ru = relu((src - out) @ TC[h]^T) -------------------------------
    for (int h = 0; h < HH; ++h) {
        diff_bf16<<<(LB * DD) / 256, 256, 0, stream>>>(
            src, out32 + (size_t)h * LB * DD, diffb + (size_t)h * LB * DD, LB * DD);
        launch_gemm(stream, diffb + (size_t)h * LB * DD, DD,
                    TCb + (size_t)h * DD * DD, DD,
                    ru32 + (size_t)h * LB * DD, DD,
                    rub + (size_t)h * LB * DD, DD,
                    nullptr, nullptr, 0, LB, DD, DD, FLAG_RELU);
    }

    // ---- per-head TransNonlinear: LN(ru + W2 relu(W1 ru + b1) + b2) ------
    for (int h = 0; h < HH; ++h) {
        launch_gemm(stream, rub + (size_t)h * LB * DD, DD,
                    ENw1b + (size_t)h * FFD * DD, DD,
                    nullptr, 0, hidb, FFD,
                    ENb1 + (size_t)h * FFD, nullptr, 0, LB, FFD, DD, FLAG_RELU);
        launch_gemm(stream, hidb, FFD,
                    ENw2b + (size_t)h * DD * FFD, FFD,
                    sum32, DD, nullptr, 0,
                    ENb2 + (size_t)h * DD,
                    ru32 + (size_t)h * LB * DD, DD, LB, DD, FFD, 0);
        // LN -> bf16 directly into concat buffer at column block h*D
        layernorm256<<<LB, 256, 0, stream>>>(sum32, DD,
                                             ENg + (size_t)h * DD,
                                             ENbeta + (size_t)h * DD,
                                             nullptr, 0,
                                             concb + (size_t)h * DD,
                                             (long long)HH * DD);
    }

    // ---- attn = concat @ Wout^T ; x = LN(src + attn) ---------------------
    launch_gemm(stream, concb, (long long)HH * DD, Woutb, (long long)HH * DD,
                sum32, DD, nullptr, 0, nullptr, src, DD, LB, DD, HH * DD, 0);
    layernorm256<<<LB, 256, 0, stream>>>(sum32, DD, N1g, N1b, x32, DD, xb, DD);

    // ---- ff = relu(x W1^T + b1) W2^T + b2 ; out = LN(x + ff) -------------
    launch_gemm(stream, xb, DD, L1wb, DD, nullptr, 0, hidb, FFD,
                L1b, nullptr, 0, LB, FFD, DD, FLAG_RELU);
    launch_gemm(stream, hidb, FFD, L2wb, FFD, sum32, DD, nullptr, 0,
                L2b, x32, DD, LB, DD, FFD, 0);
    layernorm256<<<LB, 256, 0, stream>>>(sum32, DD, N2g, N2b,
                                         outp, DD, nullptr, 0);
}
